// TtSwinSelfAttention_25271587569665
// MI455X (gfx1250) — matrix operand order repaired and checked by reading, not compiled, not run on hardware
//
#include <hip/hip_runtime.h>

// ---------------------------------------------------------------------------
// Types for CDNA5 WMMA (gfx1250, wave32)
// ---------------------------------------------------------------------------
typedef __attribute__((ext_vector_type(16))) __bf16 v16bf;
typedef __attribute__((ext_vector_type(8)))  __bf16 v8bf;
typedef __attribute__((ext_vector_type(8)))  float  v8f;

union FragBF {            // 16 bf16 operand fragment = two 16-byte chunks
    v16bf v;
    v8bf  h[2];
};

static __device__ inline v8f vzero8() {
    v8f z;
#pragma unroll
    for (int i = 0; i < 8; ++i) z[i] = 0.0f;
    return z;
}

// CDNA5 async global->LDS copy of 16 bytes (tracked with ASYNCcnt).
static __device__ inline void async_copy_b128(unsigned lds_off, const void* gaddr) {
    asm volatile("global_load_async_to_lds_b128 %0, %1, off"
                 :: "v"(lds_off), "v"((unsigned long)gaddr)
                 : "memory");
}
static __device__ inline void wait_asynccnt0() {
    asm volatile("s_wait_asynccnt 0x0" ::: "memory");
}

// ---------------------------------------------------------------------------
// Problem constants
// ---------------------------------------------------------------------------
#define HEADS   16
#define HDIM    32
#define WIN     7
#define NTOK    49        // tokens per window
#define NPAD    64        // padded tokens
#define MW      64        // windows per image
#define CDIM    512
#define BWIN    2048      // total windows
#define MROWS   (BWIN * NTOK)   // 100352 GEMM rows

// LDS row strides (bf16 elements); byte strides are multiples of 16
#define QK_LD   48        // 96 B
#define V_LD    80        // 160 B
#define P_LD    80        // 160 B

// ---------------------------------------------------------------------------
// Kernel 0: f32 -> bf16 bulk convert (8 elements / thread, b128 in & out)
// ---------------------------------------------------------------------------
__global__ __launch_bounds__(256) void cvt_f32_bf16(const float* __restrict__ src,
                                                    __bf16* __restrict__ dst,
                                                    int n8) {
    int t = blockIdx.x * blockDim.x + threadIdx.x;
    if (t >= n8) return;
    const float4* s = (const float4*)src;
    float4 a = s[2 * t + 0];
    float4 b = s[2 * t + 1];
    union { __bf16 h[8]; uint4 q; } u;
    u.h[0] = (__bf16)a.x; u.h[1] = (__bf16)a.y;
    u.h[2] = (__bf16)a.z; u.h[3] = (__bf16)a.w;
    u.h[4] = (__bf16)b.x; u.h[5] = (__bf16)b.y;
    u.h[6] = (__bf16)b.z; u.h[7] = (__bf16)b.w;
    ((uint4*)dst)[t] = u.q;
}

// ---------------------------------------------------------------------------
// Kernel 1: QKV projection GEMM  Y = X(bf16) @ W^T(bf16) + bias, bf16 out.
// grid = (3136, 1, 3): block tile = 32 rows x 512 cols (full width) so the
// activation matrix is streamed from HBM exactly once. 8 waves arranged
// 2(M) x 4(N); each wave computes 16 x 128 = 8 WMMA accumulators, so each
// A-fragment load feeds 8 v_wmma ops. Weights (1.5 MB) stay L2-resident.
// Q output pre-scaled by 1/sqrt(HD).
// ---------------------------------------------------------------------------
__global__ __launch_bounds__(256) void qkv_gemm(const __bf16* __restrict__ X,
                                                const __bf16* __restrict__ W,
                                                const float* __restrict__ bq,
                                                const float* __restrict__ bk,
                                                const float* __restrict__ bv,
                                                __bf16* __restrict__ Q,
                                                __bf16* __restrict__ K,
                                                __bf16* __restrict__ V) {
    const int p    = blockIdx.z;               // 0=q 1=k 2=v
    const int tid  = threadIdx.x;
    const int wave = tid >> 5;
    const int lane = tid & 31;
    const int l16  = lane & 15;
    const int hi   = lane >> 4;                // 0: lanes 0-15, 1: lanes 16-31

    const long mBase = (long)blockIdx.x * 32 + (wave & 1) * 16;
    const int  nBase = (wave >> 1) * 128;      // 4 N-groups cover all 512 cols

    const __bf16* Wp   = W + (long)p * CDIM * CDIM;
    const float*  bias = (p == 0) ? bq : (p == 1 ? bk : bv);
    __bf16*       dst  = (p == 0) ? Q : (p == 1 ? K : V);

    v8f acc[8];
#pragma unroll
    for (int nt = 0; nt < 8; ++nt) acc[nt] = vzero8();

    const long arow = (mBase + l16) * CDIM;
#pragma unroll 2
    for (int kt = 0; kt < 16; ++kt) {
        const int k0 = kt * 32;
        // A fragment: row = mBase + l16; K chunks {0..7,16..23} (lo) / {8..15,24..31} (hi)
        FragBF a;
        a.h[0] = *(const v8bf*)(X + arow + k0 + hi * 8);
        a.h[1] = *(const v8bf*)(X + arow + k0 + hi * 8 + 16);
#pragma unroll
        for (int nt = 0; nt < 8; ++nt) {
            // B fragment: col = nBase+nt*16+l16; K chunks {0..7,16..23} (lo) / {8..15,24..31} (hi)
            const __bf16* wptr = Wp + (long)(nBase + nt * 16 + l16) * CDIM + k0 + hi * 8;
            FragBF bfr;
            bfr.h[0] = *(const v8bf*)(wptr);
            bfr.h[1] = *(const v8bf*)(wptr + 16);
            acc[nt] = __builtin_amdgcn_wmma_f32_16x16x32_bf16(
                false, a.v, false, bfr.v, (short)0, acc[nt], false, false);
        }
    }

    const float scale = (p == 0) ? 0.17677669529663687f : 1.0f; // 1/sqrt(32)
#pragma unroll
    for (int nt = 0; nt < 8; ++nt) {
        const int   col = nBase + nt * 16 + l16;
        const float bv_ = bias[col];
#pragma unroll
        for (int i = 0; i < 8; ++i) {
            const long row = mBase + i + hi * 8;
            const float y  = (acc[nt][i] + bv_) * scale;
            dst[row * CDIM + col] = (__bf16)y;
        }
    }
}

// ---------------------------------------------------------------------------
// Kernel 2: windowed attention. One block (128 thr = 4 waves) per (window, head).
// Q/K staged to LDS via CDNA5 async global->LDS b128 copies (ASYNCcnt).
// ---------------------------------------------------------------------------
__global__ __launch_bounds__(128) void swin_attn(const __bf16* __restrict__ Q,
                                                 const __bf16* __restrict__ K,
                                                 const __bf16* __restrict__ V,
                                                 const float* __restrict__ bias_table,
                                                 const float* __restrict__ amask,
                                                 float* __restrict__ out) {
    __shared__ __align__(16) __bf16 Qs[NPAD * QK_LD];
    __shared__ __align__(16) __bf16 Ks[NPAD * QK_LD];
    __shared__ __align__(16) __bf16 Vt[HDIM * V_LD];   // V transposed [d][tok]
    __shared__ __align__(16) __bf16 Ps[NPAD * P_LD];   // probs bf16

    const int bh = blockIdx.x;
    const int b  = bh >> 4;
    const int h  = bh & 15;
    const int tid = threadIdx.x;

    const long gbase = (long)b * NTOK * CDIM + h * HDIM;

    // ---- zero-fill pad rows (tok 49..63) with regular DS stores ----
    for (int e = tid; e < (NPAD - NTOK) * HDIM; e += 128) {   // 480 elems
        const int tok = NTOK + (e >> 5), d = e & 31;
        Qs[tok * QK_LD + d] = (__bf16)0.0f;
        Ks[tok * QK_LD + d] = (__bf16)0.0f;
    }
    for (int e = tid; e < HDIM * (NPAD - NTOK); e += 128) {   // 480 elems
        const int d = e / (NPAD - NTOK), tok = NTOK + e % (NPAD - NTOK);
        Vt[d * V_LD + tok] = (__bf16)0.0f;
    }

    // ---- async-copy real Q/K rows: 49 rows x 4 chunks of 16 B each ----
    for (int c = tid; c < NTOK * 4; c += 128) {               // 196 chunks
        const int tok = c >> 2, part = c & 3;                 // part: 8-elem chunk
        const __bf16* gq = Q + gbase + (long)tok * CDIM + part * 8;
        const __bf16* gk = K + gbase + (long)tok * CDIM + part * 8;
        const unsigned lq = (unsigned)(size_t)&Qs[tok * QK_LD + part * 8];
        const unsigned lk = (unsigned)(size_t)&Ks[tok * QK_LD + part * 8];
        async_copy_b128(lq, gq);
        async_copy_b128(lk, gk);
    }

    // ---- V staged transposed through VGPRs ----
    for (int e = tid; e < HDIM * NPAD; e += 128) {
        const int d = e >> 6, tok = e & 63;
        __bf16 vv = (__bf16)0.0f;
        if (tok < NTOK) vv = V[gbase + (long)tok * CDIM + d];
        Vt[d * V_LD + tok] = vv;
    }

    wait_asynccnt0();
    __syncthreads();

    const int wave = tid >> 5;
    const int lane = tid & 31;
    const int l16  = lane & 15;
    const int hi   = lane >> 4;
    const int rowBase = wave * 16;          // 16-row strip per wave

    // ---- scores: Q (already 1/sqrt(d) scaled) @ K^T, 4 tiles of 16x16 ----
    FragBF a;
    {
        const __bf16* qp = Qs + (rowBase + l16) * QK_LD + hi * 8;
        a.h[0] = *(const v8bf*)qp;
        a.h[1] = *(const v8bf*)(qp + 16);
    }
    v8f sc[4];
#pragma unroll
    for (int t = 0; t < 4; ++t) {
        const __bf16* kp = Ks + (t * 16 + l16) * QK_LD + hi * 8;
        FragBF bb;
        bb.h[0] = *(const v8bf*)kp;
        bb.h[1] = *(const v8bf*)(kp + 16);
        sc[t] = __builtin_amdgcn_wmma_f32_16x16x32_bf16(
            false, a.v, false, bb.v, (short)0, vzero8(), false, false);
    }

    // ---- +rel-pos bias, +mask, row softmax (16-lane shfl reductions) ----
    const float* maskw = amask + (long)(b & (MW - 1)) * (NTOK * NTOK);
#pragma unroll
    for (int i = 0; i < 8; ++i) {
        const int n = rowBase + i + hi * 8;      // query token (row)
        const int i1 = n / WIN, j1 = n % WIN;
        float sv[4];
        float rmax = -3.0e38f;
#pragma unroll
        for (int t = 0; t < 4; ++t) {
            const int m = t * 16 + l16;          // key token (col)
            float val = -3.0e38f;
            if (n < NTOK && m < NTOK) {
                const int i2 = m / WIN, j2 = m % WIN;
                const int ridx = (i1 - i2 + WIN - 1) * (2 * WIN - 1) + (j1 - j2 + WIN - 1);
                val = sc[t][i] + bias_table[ridx * HEADS + h] + maskw[n * NTOK + m];
            }
            sv[t] = val;
            rmax = fmaxf(rmax, val);
        }
#pragma unroll
        for (int off = 1; off < 16; off <<= 1)
            rmax = fmaxf(rmax, __shfl_xor(rmax, off, 32));
        float rsum = 0.0f;
#pragma unroll
        for (int t = 0; t < 4; ++t) {
            const float e = (sv[t] > -1.0e38f) ? __expf(sv[t] - rmax) : 0.0f;
            sv[t] = e;
            rsum += e;
        }
#pragma unroll
        for (int off = 1; off < 16; off <<= 1)
            rsum += __shfl_xor(rsum, off, 32);
        const float inv = 1.0f / fmaxf(rsum, 1.0e-30f);
#pragma unroll
        for (int t = 0; t < 4; ++t)
            Ps[n * P_LD + t * 16 + l16] = (__bf16)(sv[t] * inv);
    }
    __syncthreads();

    // ---- ctx = P @ V : 2 d-tiles x 2 k-steps of wmma ----
#pragma unroll
    for (int dt = 0; dt < 2; ++dt) {
        v8f cacc = vzero8();
#pragma unroll
        for (int kt = 0; kt < 2; ++kt) {
            const __bf16* pp = Ps + (rowBase + l16) * P_LD + kt * 32 + hi * 8;
            FragBF pa;
            pa.h[0] = *(const v8bf*)pp;
            pa.h[1] = *(const v8bf*)(pp + 16);
            const __bf16* vp = Vt + (dt * 16 + l16) * V_LD + kt * 32 + hi * 8;
            FragBF vb;
            vb.h[0] = *(const v8bf*)vp;
            vb.h[1] = *(const v8bf*)(vp + 16);
            cacc = __builtin_amdgcn_wmma_f32_16x16x32_bf16(
                false, pa.v, false, vb.v, (short)0, cacc, false, false);
        }
#pragma unroll
        for (int i = 0; i < 8; ++i) {
            const int n = rowBase + i + hi * 8;
            if (n < NTOK)
                out[(long)(b * NTOK + n) * CDIM + h * HDIM + dt * 16 + l16] = cacc[i];
        }
    }
}

// ---------------------------------------------------------------------------
// Host launcher
// ---------------------------------------------------------------------------
extern "C" void kernel_launch(void* const* d_in, const int* in_sizes, int n_in,
                              void* d_out, int out_size, void* d_ws, size_t ws_size,
                              hipStream_t stream) {
    (void)in_sizes; (void)n_in; (void)out_size; (void)ws_size;

    const float* hs    = (const float*)d_in[0];
    const float* amask = (const float*)d_in[1];
    const float* wq    = (const float*)d_in[2];
    const float* bq    = (const float*)d_in[3];
    const float* wk    = (const float*)d_in[4];
    const float* bk    = (const float*)d_in[5];
    const float* wv    = (const float*)d_in[6];
    const float* bv    = (const float*)d_in[7];
    const float* btab  = (const float*)d_in[8];
    float* out = (float*)d_out;

    // workspace carve-up (all bf16 buffers)
    const size_t SX = (size_t)MROWS * CDIM * sizeof(short);      // 102,760,448 B
    char* ws = (char*)d_ws;
    __bf16* Xbf = (__bf16*)(ws);
    __bf16* Qbf = (__bf16*)(ws + SX);
    __bf16* Kbf = (__bf16*)(ws + 2 * SX);
    __bf16* Vbf = (__bf16*)(ws + 3 * SX);
    __bf16* Wbf = (__bf16*)(ws + 4 * SX);

    // 1) convert activations + weights to bf16
    {
        const int n8x = (MROWS * CDIM) / 8;             // 6,422,528
        cvt_f32_bf16<<<n8x / 256, 256, 0, stream>>>(hs, Xbf, n8x);
        const int n8w = (CDIM * CDIM) / 8;              // 32,768
        cvt_f32_bf16<<<n8w / 256, 256, 0, stream>>>(wq, Wbf + 0L * CDIM * CDIM, n8w);
        cvt_f32_bf16<<<n8w / 256, 256, 0, stream>>>(wk, Wbf + 1L * CDIM * CDIM, n8w);
        cvt_f32_bf16<<<n8w / 256, 256, 0, stream>>>(wv, Wbf + 2L * CDIM * CDIM, n8w);
    }

    // 2) Q/K/V projections (Q pre-scaled by 1/sqrt(HD)); X streamed once.
    {
        dim3 grid(MROWS / 32, 1, 3);                    // (3136, 1, 3)
        qkv_gemm<<<grid, 256, 0, stream>>>(Xbf, Wbf, bq, bk, bv, Qbf, Kbf, Vbf);
    }

    // 3) attention: one block per (window, head)
    {
        swin_attn<<<BWIN * HEADS, 128, 0, stream>>>(Qbf, Kbf, Vbf, btab, amask, out);
    }
}
